// CausalSelfAttention_39685497815389
// MI455X (gfx1250) — matrix operand order repaired and checked by reading, not compiled, not run on hardware
//
#include <hip/hip_runtime.h>

// ---------------------------------------------------------------------------
// Problem constants
// ---------------------------------------------------------------------------
#define BB 2
#define TT 2048
#define CC 1024
#define HH 16
#define NN 64
#define MM (BB*TT)        // 4096 rows
#define DLORA 32
#define KCHUNK 256        // K-chunk staged in LDS by the TDM per iteration

typedef unsigned short u16;
typedef __attribute__((ext_vector_type(16))) __bf16 v16bf;
typedef __attribute__((ext_vector_type(8)))  float  v8f;
typedef __attribute__((ext_vector_type(4)))  unsigned v4u;
typedef __attribute__((ext_vector_type(8)))  int      v8i;
typedef __attribute__((ext_vector_type(4)))  int      v4i;

union U16x16 { uint4 q[2]; v16bf v; };

__device__ __forceinline__ u16 f2bf(float f) {
    union { float f; unsigned u; } c; c.f = f;
    unsigned r = c.u + 0x7FFFu + ((c.u >> 16) & 1u);   // RNE
    return (u16)(r >> 16);
}

__device__ __forceinline__ float redmax16(float v) {
#pragma unroll
    for (int m = 1; m <= 8; m <<= 1) v = fmaxf(v, __shfl_xor(v, m, 32));
    return v;
}
__device__ __forceinline__ float redsum16(float v) {
#pragma unroll
    for (int m = 1; m <= 8; m <<= 1) v += __shfl_xor(v, m, 32);
    return v;
}
__device__ __forceinline__ float redsum32(float v) {
#pragma unroll
    for (int m = 1; m <= 16; m <<= 1) v += __shfl_xor(v, m, 32);
    return v;
}

// ---------------------------------------------------------------------------
// fp32 -> bf16 conversion
// ---------------------------------------------------------------------------
__global__ __launch_bounds__(256) void k_convert_bf16(const float* __restrict__ src,
                                                      u16* __restrict__ dst, int n) {
    int i = blockIdx.x * 256 + threadIdx.x;
    if (i < n) dst[i] = f2bf(src[i]);
}

// ---------------------------------------------------------------------------
// TDM helper: issue a Tensor-DMA load of a 64-row x KCHUNK-col bf16 tile of
// Bw (row stride K elements) into LDS byte offset ldsoff.
// D# layout per CDNA5 ISA 8.3/8.4 (group0 128b, group1 256b; groups 2/3 zero
// for a 2D tile).  6-arg builtin form (clang-23 / therock-10.0 lane).
// ---------------------------------------------------------------------------
__device__ __forceinline__ void tdm_load_b_tile(const u16* Bw, int K, int n0,
                                                int c, unsigned ldsoff) {
    unsigned long long ga =
        (unsigned long long)(uintptr_t)(Bw + (size_t)n0 * K + (size_t)c * KCHUNK);
    v4u g0;
    g0[0] = 1u;                                            // count=1 (valid), user mode
    g0[1] = ldsoff;                                        // lds_addr (bytes)
    g0[2] = (unsigned)(ga & 0xffffffffu);                  // global_addr[31:0]
    g0[3] = (unsigned)((ga >> 32) & 0x01ffffffu)           // global_addr[56:32]
          | (2u << 30);                                    // type=2 ("image")
    v8i g1;
    g1[0] = (int)(1u << 16);                               // wg_mask=0, data_size=1 (2B)
    g1[1] = (int)((unsigned)(KCHUNK & 0xffff) << 16);      // tensor_dim0[15:0]
    g1[2] = (int)((unsigned)(64  & 0xffff) << 16);         // dim0 hi=0 | tensor_dim1 lo
    g1[3] = (int)((unsigned)(KCHUNK & 0xffff) << 16);      // dim1 hi=0 | tile_dim0
    g1[4] = 64;                                            // tile_dim1=64, tile_dim2=0
    g1[5] = K;                                             // tensor_dim0_stride[31:0]
    g1[6] = 0;                                             // stride0 hi | stride1 lo
    g1[7] = 0;                                             // stride1 hi
    v4i z4 = {0, 0, 0, 0};                                 // groups 2/3 unused (2D)
    v8i z8 = {0, 0, 0, 0, 0, 0, 0, 0};
    __builtin_amdgcn_tensor_load_to_lds(g0, g1, z4, z4, z8, 0);
}

// ---------------------------------------------------------------------------
// WMMA GEMM with TDM-staged weights:
//   out[M,Nout] = A[M,K] @ Bw[Nout,K]^T  (+ optional add)
// Block = 4 waves = 128(M) x 64(N) tile; each wave owns 32 M-rows.
// Wave 0 drives the Tensor Data Mover: double-buffered 64x256 bf16 weight
// tiles (32 KB each) are DMA'd to LDS (TENSORcnt), overlapped with compute.
// A fragments load straight from global in contiguous k order (k = 16*hsel + s); B
// fragments come from LDS as two ds_load_b128 per fragment.
// ---------------------------------------------------------------------------
__global__ __launch_bounds__(128) void k_gemm_tdm(const u16* __restrict__ A,
                                                  const u16* __restrict__ Bw,
                                                  const float* __restrict__ add,
                                                  float* __restrict__ out,
                                                  int M, int K, int Nout, int ldo) {
    __shared__ __align__(16) u16 bsh[2][64 * KCHUNK];      // 2 x 32 KB
    const int lane   = threadIdx.x & 31;
    const int wave   = threadIdx.x >> 5;
    const int lane16 = lane & 15;
    const int hsel   = lane >> 4;
    const int m0 = blockIdx.x * 128 + wave * 32;
    const int n0 = blockIdx.y * 64;
    const int nchunk = K / KCHUNK;

    v8f acc[2][4] = {};

    // prologue: wave 0 DMAs chunk 0, everyone waits
    if (wave == 0) {
        tdm_load_b_tile(Bw, K, n0, 0, 0u);
        __builtin_amdgcn_s_wait_tensorcnt(0);
    }
    __syncthreads();

    for (int c = 0; c < nchunk; ++c) {
        // overlap: kick off next chunk's TDM while computing this one
        if (wave == 0 && c + 1 < nchunk)
            tdm_load_b_tile(Bw, K, n0, c + 1,
                            (unsigned)(((c + 1) & 1) * (64 * KCHUNK * 2)));

        const u16* bs = bsh[c & 1];
        // The TDM writes LDS behind the compiler's back (no visible IR store
        // to bsh).  This opaque asm is treated as an unknown memory writer so
        // the B-tile ds_load_b128s below cannot be folded away or hoisted.
        asm volatile("" :: "r"(bs) : "memory");

        for (int kk = 0; kk < KCHUNK; kk += 32) {
            U16x16 a[2], b[4];
#pragma unroll
            for (int i = 0; i < 2; ++i) {
                const u16* ap = A + (size_t)(m0 + i*16 + lane16) * K
                                  + c * KCHUNK + kk + hsel*16;
                a[i].q[0] = *(const uint4*)ap;
                a[i].q[1] = *(const uint4*)(ap + 8);
            }
#pragma unroll
            for (int j = 0; j < 4; ++j) {
                const u16* bp = bs + (j*16 + lane16) * KCHUNK + kk + hsel*16;
                b[j].q[0] = *(const uint4*)bp;
                b[j].q[1] = *(const uint4*)(bp + 8);
            }
#pragma unroll
            for (int i = 0; i < 2; ++i)
#pragma unroll
                for (int j = 0; j < 4; ++j)
                    acc[i][j] = __builtin_amdgcn_wmma_f32_16x16x32_bf16(
                        false, a[i].v, false, b[j].v, (short)0, acc[i][j], false, false);
        }

        if (wave == 0 && c + 1 < nchunk)
            __builtin_amdgcn_s_wait_tensorcnt(0);        // chunk c+1 landed in LDS
        __syncthreads();
    }

#pragma unroll
    for (int i = 0; i < 2; ++i)
#pragma unroll
        for (int j = 0; j < 4; ++j)
#pragma unroll
            for (int r = 0; r < 8; ++r) {
                int row = m0 + i*16 + r + 8*hsel;
                int col = n0 + j*16 + lane16;
                float v = acc[i][j][r];
                if (add) v += add[(size_t)row * ldo + col];
                out[(size_t)row * ldo + col] = v;
            }
}

// ---------------------------------------------------------------------------
// LoRA stage 1: t[M,32] = x[M,C] @ lora_a[C,32]
// ---------------------------------------------------------------------------
__global__ __launch_bounds__(256) void k_lora1(const float* __restrict__ x,
                                               const float* __restrict__ la,
                                               float* __restrict__ t) {
    int idx = blockIdx.x * 256 + threadIdx.x;          // M*32 threads
    int m = idx >> 5, d = idx & 31;
    float s = 0.f;
    const float* xr = x + (size_t)m * CC;
    for (int k = 0; k < CC; ++k) s += xr[k] * la[k * DLORA + d];
    t[idx] = s;
}

// LoRA stage 2 + value-residual gate: v = v + (v1 - v)*sigmoid(v0 + t@lora_b)
// qkv buffer layout: [M, 3072]; v starts at column 2048.
__global__ __launch_bounds__(256) void k_lora2(const float* __restrict__ t,
                                               const float* __restrict__ lb,
                                               const float* __restrict__ v0,
                                               const float* __restrict__ v1,
                                               float* __restrict__ qkv) {
    int idx = blockIdx.x * 256 + threadIdx.x;          // M*C threads
    int m = idx >> 10, c = idx & 1023;
    float z = v0[c];
    const float* tr = t + m * DLORA;
#pragma unroll
    for (int d = 0; d < DLORA; ++d) z += tr[d] * lb[d * CC + c];
    float g = 1.f / (1.f + __expf(-z));
    float* vp = qkv + (size_t)m * 3072 + 2048 + c;
    float vv = *vp;
    *vp = vv + (v1[(size_t)m * CC + c] - vv) * g;
}

// ---------------------------------------------------------------------------
// Prep: token-shift (+ per-head LN + partial RoPE for q/k), bf16 repack.
// One wave handles one (b,t,h); lane l owns channels n=2l, 2l+1.
// q/k out: [B,H,T,64] bf16 ; v out (transpose=1): [B,H,64,T] bf16
// ---------------------------------------------------------------------------
__global__ __launch_bounds__(256) void k_prep(const float* __restrict__ qkv, int off,
                                              const float* __restrict__ mix,
                                              const float* __restrict__ lnw,
                                              const float* __restrict__ lnb,
                                              int do_lnrope, int transpose_out,
                                              u16* __restrict__ out) {
    int gid  = blockIdx.x * 256 + threadIdx.x;
    int widx = gid >> 5;                               // 0 .. B*T*H-1
    int lane = gid & 31;
    int h = widx % HH;
    int t = (widx / HH) % TT;
    int b = widx / (HH * TT);
    int n0 = 2 * lane;

    const float* cur = qkv + (size_t)(b*TT + t)              * 3072 + off + h*NN;
    const float* prv = qkv + (size_t)(b*TT + (t ? t-1 : 0))  * 3072 + off + h*NN;
    float mg0 = mix[h*NN + n0], mg1 = mix[h*NN + n0 + 1];
    float s0 = cur[n0]     + mg0 * (prv[n0]     - cur[n0]);
    float s1 = cur[n0 + 1] + mg1 * (prv[n0 + 1] - cur[n0 + 1]);

    if (do_lnrope) {
        float mu = redsum32(s0 + s1) * (1.f / NN);
        float d0 = s0 - mu, d1 = s1 - mu;
        float var = redsum32(d0*d0 + d1*d1) * (1.f / NN);
        float rstd = rsqrtf(var + 1e-5f);
        s0 = d0 * rstd * lnw[n0]     + lnb[n0];
        s1 = d1 * rstd * lnw[n0 + 1] + lnb[n0 + 1];
        if (n0 < 32) {                                  // partial RoPE, pair idx = lane
            float e   = (float)lane * (1.0f / 15.0f);   // linspace(0,1,16)
            float ang = __expf(-e * __logf(10000.0f));
            float th  = (float)t * ang;
            float cs = __cosf(th), sn = __sinf(th);
            float r0 = cs * s0 + sn * s1;               // sin sign +1 at even, xf=odd
            float r1 = cs * s1 - sn * s0;               // sin sign -1 at odd,  xf=even
            s0 = r0; s1 = r1;
        }
    }

    if (!transpose_out) {
        unsigned pk = (unsigned)f2bf(s0) | ((unsigned)f2bf(s1) << 16);
        *(unsigned*)(out + ((size_t)((b*HH + h)*TT + t)) * NN + n0) = pk;
    } else {
        out[((size_t)((b*HH + h)*NN + n0    )) * TT + t] = f2bf(s0);
        out[((size_t)((b*HH + h)*NN + n0 + 1)) * TT + t] = f2bf(s1);
    }
}

// ---------------------------------------------------------------------------
// Causal flash attention with WMMA bf16.
// grid = (32 qtile-groups, B*H); block = 128 (4 waves); 1 wave = 16 queries.
// scores: QK^T via 2 chained wmma (K-dim = head dim 64) per 16-key tile.
// PV:     probs (transposed C->A layout through wave-private LDS) @ V.
// ---------------------------------------------------------------------------
__global__ __launch_bounds__(128) void k_attn(const u16* __restrict__ qh,
                                              const u16* __restrict__ kh,
                                              const u16* __restrict__ vT,
                                              u16* __restrict__ ybf) {
    __shared__ __align__(16) u16 pbuf[4][16 * 32];
    const int wave   = threadIdx.x >> 5;
    const int lane   = threadIdx.x & 31;
    const int lane16 = lane & 15;
    const int hsel   = lane >> 4;
    const int qtile  = blockIdx.x * 4 + wave;
    const int q0     = qtile * 16;
    const int bh     = blockIdx.y;
    const int b      = bh / HH, h = bh % HH;

    // Q fragments (A-layout), head-dim split in two K=32 chunks
    U16x16 aq[2];
#pragma unroll
    for (int c = 0; c < 2; ++c) {
        const u16* qp = qh + ((size_t)bh*TT + q0 + lane16) * NN + c*32 + hsel*16;
        aq[c].q[0] = *(const uint4*)qp;
        aq[c].q[1] = *(const uint4*)(qp + 8);
    }

    float mrow[8], lrow[8];
#pragma unroll
    for (int r = 0; r < 8; ++r) { mrow[r] = -INFINITY; lrow[r] = 0.f; }
    v8f o[4] = {};

    for (int kb = 0; kb < q0 + 16; kb += 32) {
        // ---- scores: two 16x16 tiles over 32 keys ----
        v8f s[2];
#pragma unroll
        for (int kt = 0; kt < 2; ++kt) {
            v8f sa = {};
#pragma unroll
            for (int c = 0; c < 2; ++c) {
                U16x16 bk;
                const u16* kp = kh + ((size_t)bh*TT + kb + kt*16 + lane16) * NN
                                   + c*32 + hsel*16;
                bk.q[0] = *(const uint4*)kp;
                bk.q[1] = *(const uint4*)(kp + 8);
                sa = __builtin_amdgcn_wmma_f32_16x16x32_bf16(
                        false, aq[c].v, false, bk.v, (short)0, sa, false, false);
            }
            s[kt] = sa;
        }

        // ---- online softmax (rows M = r + 8*hsel) ----
        float alpha[8];
#pragma unroll
        for (int r = 0; r < 8; ++r) {
            int qrow = q0 + r + 8*hsel;
            float v0s = s[0][r] * 0.125f;
            float v1s = s[1][r] * 0.125f;
            if (kb      + lane16 > qrow) v0s = -INFINITY;
            if (kb + 16 + lane16 > qrow) v1s = -INFINITY;
            float rm = redmax16(fmaxf(v0s, v1s));
            float nm = fmaxf(mrow[r], rm);
            alpha[r] = __expf(mrow[r] - nm);
            mrow[r]  = nm;
            float p0 = __expf(v0s - nm);
            float p1 = __expf(v1s - nm);
            lrow[r] = lrow[r] * alpha[r] + redsum16(p0 + p1);
            // C-layout -> LDS row-major [query][key] bf16
            pbuf[wave][(r + 8*hsel) * 32 +      lane16] = f2bf(p0);
            pbuf[wave][(r + 8*hsel) * 32 + 16 + lane16] = f2bf(p1);
        }
#pragma unroll
        for (int dt = 0; dt < 4; ++dt)
#pragma unroll
            for (int r = 0; r < 8; ++r) o[dt][r] *= alpha[r];

        asm volatile("s_wait_dscnt 0" ::: "memory");
        __builtin_amdgcn_wave_barrier();

        // probs in A-layout from LDS (two b128 ds loads per lane)
        U16x16 ap;
        ap.q[0] = *(const uint4*)&pbuf[wave][lane16 * 32 +     hsel*16];
        ap.q[1] = *(const uint4*)&pbuf[wave][lane16 * 32 + 8 + hsel*16];

        // ---- PV: 4 output d-tiles, K = 32 keys ----
#pragma unroll
        for (int dt = 0; dt < 4; ++dt) {
            U16x16 bv;
            const u16* vp = vT + ((size_t)bh*NN + dt*16 + lane16) * TT + kb + hsel*16;
            bv.q[0] = *(const uint4*)vp;
            bv.q[1] = *(const uint4*)(vp + 8);
            o[dt] = __builtin_amdgcn_wmma_f32_16x16x32_bf16(
                        false, ap.v, false, bv.v, (short)0, o[dt], false, false);
        }
    }

    // ---- finalize: divide by l, store to y as [B,T,C] bf16 for the proj GEMM ----
#pragma unroll
    for (int dt = 0; dt < 4; ++dt)
#pragma unroll
        for (int r = 0; r < 8; ++r) {
            int t   = q0 + r + 8*hsel;
            int col = h*NN + dt*16 + lane16;
            float v = o[dt][r] / lrow[r];
            ybf[((size_t)(b*TT + t)) * CC + col] = f2bf(v);
        }
}

// ---------------------------------------------------------------------------
// Launch
// ---------------------------------------------------------------------------
extern "C" void kernel_launch(void* const* d_in, const int* in_sizes, int n_in,
                              void* d_out, int out_size, void* d_ws, size_t ws_size,
                              hipStream_t stream) {
    const float* residual = (const float*)d_in[0];
    const float* x        = (const float*)d_in[1];
    const float* v1       = (const float*)d_in[2];
    const float* Wq       = (const float*)d_in[5];
    const float* Wk       = (const float*)d_in[6];
    const float* Wv       = (const float*)d_in[7];
    const float* Wproj    = (const float*)d_in[8];
    const float* lnq_w    = (const float*)d_in[9];
    const float* lnq_b    = (const float*)d_in[10];
    const float* lnk_w    = (const float*)d_in[11];
    const float* lnk_b    = (const float*)d_in[12];
    const float* xq_mix   = (const float*)d_in[13];
    const float* xk_mix   = (const float*)d_in[14];
    const float* xv_mix   = (const float*)d_in[15];
    const float* lora_a   = (const float*)d_in[16];
    const float* lora_b   = (const float*)d_in[17];
    const float* v0       = (const float*)d_in[18];
    float* out = (float*)d_out;

    char* ws = (char*)d_ws;
    size_t off = 0;
    u16*   xbf   = (u16*)(ws + off);  off += (size_t)MM * CC * 2;          // 8 MB
    u16*   wqkv  = (u16*)(ws + off);  off += (size_t)3 * CC * CC * 2;      // 6 MB
    u16*   wpbf  = (u16*)(ws + off);  off += (size_t)CC * CC * 2;          // 2 MB
    float* qkvf  = (float*)(ws + off); off += (size_t)MM * 3 * CC * 4;     // 48 MB
    float* tlora = (float*)(ws + off); off += (size_t)MM * DLORA * 4;      // 0.5 MB
    u16*   qhb   = (u16*)(ws + off);  off += (size_t)MM * CC * 2;          // 8 MB
    u16*   khb   = (u16*)(ws + off);  off += (size_t)MM * CC * 2;          // 8 MB
    u16*   vtb   = (u16*)(ws + off);  off += (size_t)MM * CC * 2;          // 8 MB
    u16*   ybf   = (u16*)(ws + off);  off += (size_t)MM * CC * 2;          // 8 MB

    const int NW = CC * CC;          // 1M elems per weight matrix
    const int NX = MM * CC;          // 4M elems

    k_convert_bf16<<<(NX + 255)/256, 256, 0, stream>>>(x,     xbf,          NX);
    k_convert_bf16<<<(NW + 255)/256, 256, 0, stream>>>(Wq,    wqkv,         NW);
    k_convert_bf16<<<(NW + 255)/256, 256, 0, stream>>>(Wk,    wqkv + NW,    NW);
    k_convert_bf16<<<(NW + 255)/256, 256, 0, stream>>>(Wv,    wqkv + 2*NW,  NW);
    k_convert_bf16<<<(NW + 255)/256, 256, 0, stream>>>(Wproj, wpbf,         NW);

    // fused QKV projection: [4096,1024] @ [3072,1024]^T -> [4096,3072]
    k_gemm_tdm<<<dim3(MM/128, 3*CC/64), 128, 0, stream>>>(
        xbf, wqkv, nullptr, qkvf, MM, CC, 3*CC, 3*CC);

    k_lora1<<<(MM*DLORA)/256, 256, 0, stream>>>(x, lora_a, tlora);
    k_lora2<<<(MM*CC)/256,    256, 0, stream>>>(tlora, lora_b, v0, v1, qkvf);

    // token-shift + LN + RoPE repacks
    k_prep<<<(BB*TT*HH*32)/256, 256, 0, stream>>>(qkvf,    0, xq_mix, lnq_w, lnq_b, 1, 0, qhb);
    k_prep<<<(BB*TT*HH*32)/256, 256, 0, stream>>>(qkvf, 1024, xk_mix, lnk_w, lnk_b, 1, 0, khb);
    k_prep<<<(BB*TT*HH*32)/256, 256, 0, stream>>>(qkvf, 2048, xv_mix, nullptr, nullptr, 0, 1, vtb);

    // causal attention
    k_attn<<<dim3(TT/16/4, BB*HH), 128, 0, stream>>>(qhb, khb, vtb, ybf);

    // output projection + residual
    k_gemm_tdm<<<dim3(MM/128, CC/64), 128, 0, stream>>>(
        ybf, wpbf, residual, out, MM, CC, CC, CC);
}
